// MLP_learner_8615704396633
// MI455X (gfx1250) — compile-verified, matrix operand order checked by reading, not run on hardware
//
#include <hip/hip_runtime.h>

#define N_ROWS 12288
#define DIM    256
#define K_KEEP 31

typedef __attribute__((ext_vector_type(16))) _Float16 v16h;
typedef __attribute__((ext_vector_type(8)))  _Float16 v8h;
typedef __attribute__((ext_vector_type(8)))  float    v8f;

// Build a 16x32 f16 WMMA fragment for lane (m = lane%16, half = lane/16).
// Per CDNA5 ISA layout: VGPR0..3 hold K = k+8*half+0..7 (contiguous),
// VGPR4..7 hold K = k+16+8*half+0..7 (contiguous) -> two 16-byte loads.
__device__ __forceinline__ v16h load_frag(const _Float16* __restrict__ rowbase,
                                          int k, int half) {
  union { v16h v; v8h h[2]; } u;
  u.h[0] = *reinterpret_cast<const v8h*>(rowbase + k + half * 8);
  u.h[1] = *reinterpret_cast<const v8h*>(rowbase + k + half * 8 + 16);
  return u.v;
}

// ---------------------------------------------------------------------------
// Kernel 1: emb = L2norm(relu(X @ W0^T + b0) @ W1^T + b1), stored as f16.
// One 256-thread block per row; weights stay hot in L2.
// ---------------------------------------------------------------------------
__global__ __launch_bounds__(256) void emb_kernel(
    const float* __restrict__ X,
    const float* __restrict__ W0, const float* __restrict__ b0,
    const float* __restrict__ W1, const float* __restrict__ b1,
    _Float16* __restrict__ embh) {
  __shared__ float xs[DIM];
  __shared__ float hs[DIM];
  __shared__ float red[8];
  const int j   = threadIdx.x;
  const int row = blockIdx.x;

  xs[j] = X[(size_t)row * DIM + j];
  __syncthreads();

  // h0[j] = relu(b0[j] + sum_k x[k] * W0[j][k])   (x @ W0^T)
  const float* w = W0 + (size_t)j * DIM;
  float s = b0[j];
  #pragma unroll 4
  for (int k = 0; k < DIM; ++k) s = fmaf(xs[k], w[k], s);
  s = fmaxf(s, 0.0f);
  hs[j] = s;
  __syncthreads();

  // h1[j] = b1[j] + sum_k h0[k] * W1[j][k]
  w = W1 + (size_t)j * DIM;
  float t = b1[j];
  #pragma unroll 4
  for (int k = 0; k < DIM; ++k) t = fmaf(hs[k], w[k], t);

  // row L2 norm (wave32 shuffle reduce + cross-wave via LDS)
  float sq = t * t;
  #pragma unroll
  for (int o = 16; o > 0; o >>= 1) sq += __shfl_xor(sq, o, 32);
  if ((j & 31) == 0) red[j >> 5] = sq;
  __syncthreads();
  float tot = 0.0f;
  #pragma unroll
  for (int i = 0; i < 8; ++i) tot += red[i];
  const float nrm = fmaxf(sqrtf(tot), 1e-12f);
  embh[(size_t)row * DIM + j] = (_Float16)(t / nrm);
}

// ---------------------------------------------------------------------------
// Kernel 2: sim = emb @ emb^T via v_wmma_f32_16x16x32_f16.
// Block = 8 waves; wave tile = 32x32 (2x2 of 16x16), block tile = 64x128.
// emb (6.3 MB f16) stays resident in L2, so all fragment loads hit cache.
// ---------------------------------------------------------------------------
__global__ __launch_bounds__(256) void sim_wmma(
    const _Float16* __restrict__ embh, float* __restrict__ out) {
  const int wave = threadIdx.x >> 5;
  const int lane = threadIdx.x & 31;
  const int half = lane >> 4;
  const int lm   = lane & 15;

  const int bm = blockIdx.x / (N_ROWS / 128);
  const int bn = blockIdx.x % (N_ROWS / 128);
  const int waveM = bm * 64  + (wave >> 2) * 32;
  const int waveN = bn * 128 + (wave & 3) * 32;

  const _Float16* a0p = embh + (size_t)(waveM + lm)      * DIM;
  const _Float16* a1p = embh + (size_t)(waveM + 16 + lm) * DIM;
  const _Float16* b0p = embh + (size_t)(waveN + lm)      * DIM;
  const _Float16* b1p = embh + (size_t)(waveN + 16 + lm) * DIM;

  v8f acc00 = {}, acc01 = {}, acc10 = {}, acc11 = {};

  #pragma unroll
  for (int k = 0; k < DIM; k += 32) {
    v16h a0 = load_frag(a0p, k, half);
    v16h a1 = load_frag(a1p, k, half);
    v16h b0 = load_frag(b0p, k, half);
    v16h b1 = load_frag(b1p, k, half);
    acc00 = __builtin_amdgcn_wmma_f32_16x16x32_f16(false, a0, false, b0,
                                                   (short)0, acc00, false, false);
    acc01 = __builtin_amdgcn_wmma_f32_16x16x32_f16(false, a0, false, b1,
                                                   (short)0, acc01, false, false);
    acc10 = __builtin_amdgcn_wmma_f32_16x16x32_f16(false, a1, false, b0,
                                                   (short)0, acc10, false, false);
    acc11 = __builtin_amdgcn_wmma_f32_16x16x32_f16(false, a1, false, b1,
                                                   (short)0, acc11, false, false);
  }

  // C/D layout: vgpr v holds (M = v + 8*half, N = lane%16)
  #pragma unroll
  for (int v = 0; v < 8; ++v) {
    const int m0 = waveM + v + 8 * half;
    const int m1 = m0 + 16;
    out[(size_t)m0 * N_ROWS + waveN + lm]      = acc00[v];
    out[(size_t)m0 * N_ROWS + waveN + 16 + lm] = acc01[v];
    out[(size_t)m1 * N_ROWS + waveN + lm]      = acc10[v];
    out[(size_t)m1 * N_ROWS + waveN + 16 + lm] = acc11[v];
  }
}

// ---------------------------------------------------------------------------
// Kernel 3: per-row 31st-largest value via binary search on order-preserving
// uint keys, done entirely in LDS after a single DRAM read of the row.
// ---------------------------------------------------------------------------
__global__ __launch_bounds__(256) void topk_threshold(
    const float* __restrict__ sim, unsigned int* __restrict__ thrkeys) {
  __shared__ unsigned int keys[N_ROWS];   // 48 KB
  __shared__ int cnt;
  const int tid = threadIdx.x;
  const size_t rowoff = (size_t)blockIdx.x * N_ROWS;

  for (int i = tid; i < N_ROWS; i += 256) {
    unsigned int b = __float_as_uint(sim[rowoff + i]);
    keys[i] = (b & 0x80000000u) ? ~b : (b | 0x80000000u);
  }
  __syncthreads();

  // invariant: count(key >= lo) >= K_KEEP, count(key >= hi) < K_KEEP
  unsigned int lo = 0u, hi = 0xFFFFFFFFu;
  while (hi - lo > 1u) {
    const unsigned int mid = lo + ((hi - lo) >> 1);
    if (tid == 0) cnt = 0;
    __syncthreads();
    int c = 0;
    for (int i = tid; i < N_ROWS; i += 256) c += (keys[i] >= mid) ? 1 : 0;
    #pragma unroll
    for (int o = 16; o > 0; o >>= 1) c += __shfl_xor(c, o, 32);
    if ((tid & 31) == 0) atomicAdd(&cnt, c);
    __syncthreads();
    if (cnt >= K_KEEP) lo = mid; else hi = mid;
    __syncthreads();
  }
  if (tid == 0) thrkeys[blockIdx.x] = lo;   // key of the 31st-largest value
}

// ---------------------------------------------------------------------------
// Kernel 4: out = (key(v) >= thr_row) ? relu(v) : 0, float4-vectorized.
// ---------------------------------------------------------------------------
__global__ __launch_bounds__(256) void apply_topk_relu(
    float* __restrict__ sim, const unsigned int* __restrict__ thrkeys) {
  const int row  = blockIdx.y;
  const int col4 = (blockIdx.x * 256 + threadIdx.x) * 4;
  const unsigned int thr = thrkeys[row];
  float4* p = reinterpret_cast<float4*>(sim + (size_t)row * N_ROWS + col4);
  float4 v = *p;
  auto keep = [&](float x) {
    unsigned int b = __float_as_uint(x);
    unsigned int key = (b & 0x80000000u) ? ~b : (b | 0x80000000u);
    return (key >= thr) ? fmaxf(x, 0.0f) : 0.0f;
  };
  v.x = keep(v.x); v.y = keep(v.y); v.z = keep(v.z); v.w = keep(v.w);
  *p = v;
}

// ---------------------------------------------------------------------------
extern "C" void kernel_launch(void* const* d_in, const int* in_sizes, int n_in,
                              void* d_out, int out_size, void* d_ws, size_t ws_size,
                              hipStream_t stream) {
  const float* X  = (const float*)d_in[0];
  const float* W0 = (const float*)d_in[1];
  const float* b0 = (const float*)d_in[2];
  const float* W1 = (const float*)d_in[3];
  const float* b1 = (const float*)d_in[4];
  float* out = (float*)d_out;

  // workspace layout: [emb f16: 12288*256*2 B][thresholds: 12288*4 B]
  _Float16* embh = (_Float16*)d_ws;
  unsigned int* thrkeys =
      (unsigned int*)((char*)d_ws + (size_t)N_ROWS * DIM * sizeof(_Float16));

  emb_kernel<<<N_ROWS, 256, 0, stream>>>(X, W0, b0, W1, b1, embh);
  sim_wmma<<<(N_ROWS / 64) * (N_ROWS / 128), 256, 0, stream>>>(embh, out);
  topk_threshold<<<N_ROWS, 256, 0, stream>>>(out, thrkeys);
  dim3 g4(N_ROWS / 1024, N_ROWS);
  apply_topk_relu<<<g4, 256, 0, stream>>>(out, thrkeys);
}
